// MultiHeadSlotAttention_50663434223981
// MI455X (gfx1250) — matrix operand order repaired and checked by reading, not compiled, not run on hardware
//
#include <hip/hip_runtime.h>
#include <cstdint>
#include <cstddef>

// ---------------- types ----------------
typedef __bf16 bf16_t;
typedef __attribute__((ext_vector_type(16))) __bf16 v16bf;
typedef __attribute__((ext_vector_type(8)))  __bf16 v8bf;
typedef __attribute__((ext_vector_type(8)))  float  v8f;
typedef __attribute__((ext_vector_type(4)))  unsigned int u32x4;
typedef __attribute__((ext_vector_type(8)))  int i32x8;
typedef __attribute__((ext_vector_type(4)))  int i32x4;

union BF16x16 { v16bf v; v8bf h[2]; };

// ---------------- problem constants ----------------
#define Bc   16
#define Nc   8192
#define DINc 768
#define Dc   768
#define Hc   12
#define HDc  64
#define Sc   8
#define ITERSc 3
#define EPSc   1e-8f
#define LNEPSc 1e-5f
#define SCALEc 0.036084391824351614f   // 768^-0.5

#define BN   (Bc*Nc)                    // 131072 tokens

// ---------------- workspace layout (bytes) ----------------
#define OFF_XN   ((size_t)0)                                   // bf16 [BN, DIN]
#define OFF_K    (OFF_XN + (size_t)BN*DINc*2)                  // bf16 [BN, D]
#define OFF_V    (OFF_K  + (size_t)BN*Dc*2)                    // bf16 [BN, D]
#define OFF_Q    (OFF_V  + (size_t)BN*Dc*2)                    // bf16 [B*S, D]
#define OFF_UPD  (OFF_Q  + (size_t)Bc*Sc*Dc*2)                 // f32  [B,H,S,HD]
#define OFF_RS   (OFF_UPD+ (size_t)Bc*Hc*Sc*HDc*4)             // f32  [B,H,S]
#define OFF_VSUM (OFF_RS + (size_t)Bc*Hc*Sc*4)                 // f32  [B,D]
#define OFF_WKB  (OFF_VSUM + (size_t)Bc*Dc*4)                  // bf16 [D, DIN]
#define OFF_WVB  (OFF_WKB + (size_t)Dc*DINc*2)                 // bf16 [D, DIN]

// ---------------- Tensor Data Mover support ----------------
#if defined(__has_builtin)
#  if __has_builtin(__builtin_amdgcn_tensor_load_to_lds)
#    define HAVE_TDM 1
#  endif
#endif

#ifdef HAVE_TDM
// Issue a TDM DMA of a 128(row) x 32(col) bf16 tile, row stride 768 elements,
// from global `gptr` (tile start) into LDS at byte address `ldsAddr`.
// D# per CDNA5 ISA: group0 {count,lds,global,type=2}; group1 {dims,tile,strides}.
__device__ __forceinline__ void tdm_load_tile(const bf16_t* gptr, unsigned ldsAddr) {
  unsigned long long ga = (unsigned long long)(uintptr_t)gptr;
  u32x4 g0;
  g0[0] = 1u;                                   // count=1, user descriptor
  g0[1] = ldsAddr;                              // lds_addr[31:0]
  g0[2] = (unsigned)ga;                         // global_addr[31:0]
  g0[3] = ((unsigned)(ga >> 32) & 0x01FFFFFFu)  // global_addr[56:32]
          | (2u << 30);                         // type=2 (image)
  i32x8 g1;
  g1[0] = (int)(1u << 16);                      // workgroup_mask=0, data_size=1 (2B)
  g1[1] = (int)(768u << 16);                    // tensor_dim0[15:0]=768 at bits 63:48
  g1[2] = 0;                                    // tensor_dim0 hi, tensor_dim1 lo (BN lo16=0)
  g1[3] = (int)((BN >> 16) | (32u << 16));      // tensor_dim1 hi =2, tile_dim0=32
  g1[4] = 128;                                  // tile_dim1=128, tile_dim2=0
  g1[5] = 768;                                  // tensor_dim0_stride[31:0]=768
  g1[6] = 0;
  g1[7] = 0;
  i32x4 gz; gz[0] = 0; gz[1] = 0; gz[2] = 0; gz[3] = 0;   // 2-D tensor: groups 2/3 unused
#if defined(__clang_major__) && (__clang_major__ >= 23)
  i32x8 g4;
  g4[0]=0; g4[1]=0; g4[2]=0; g4[3]=0; g4[4]=0; g4[5]=0; g4[6]=0; g4[7]=0;
  __builtin_amdgcn_tensor_load_to_lds(g0, g1, gz, gz, g4, 0);
#else
  __builtin_amdgcn_tensor_load_to_lds(g0, g1, gz, gz, 0);
#endif
}
#endif

// =====================================================================
// K_init: broadcast slots_init -> slots (d_out)
// =====================================================================
__global__ __launch_bounds__(256) void init_slots_kernel(
    const float* __restrict__ slots_init, float* __restrict__ slots) {
  int i = blockIdx.x * 256 + threadIdx.x;
  if (i < Bc*Sc*Dc) slots[i] = slots_init[i % (Sc*Dc)];
}

// =====================================================================
// K_wcvt: fp32 weights -> bf16 (one shot)
// =====================================================================
__global__ __launch_bounds__(256) void wcvt_kernel(
    const float* __restrict__ W, bf16_t* __restrict__ out, int n) {
  int i = blockIdx.x * 256 + threadIdx.x;
  if (i < n) out[i] = (bf16_t)W[i];
}

// =====================================================================
// K0: LayerNorm(inputs) -> xn bf16. One wave per token.
// =====================================================================
__global__ __launch_bounds__(256) void ln_in_kernel(
    const float* __restrict__ x, const float* __restrict__ g,
    const float* __restrict__ beta, bf16_t* __restrict__ xn) {
  int wave = threadIdx.x >> 5, lane = threadIdx.x & 31;
  int tok = blockIdx.x * 8 + wave;
  const float* row = x + (size_t)tok * DINc;
  float ps = 0.f, pq = 0.f;
  for (int d = lane; d < DINc; d += 32) { float v = row[d]; ps += v; pq += v*v; }
  for (int m = 16; m >= 1; m >>= 1) { ps += __shfl_xor(ps, m, 32); pq += __shfl_xor(pq, m, 32); }
  float mean = ps * (1.f/DINc);
  float var  = pq * (1.f/DINc) - mean*mean;
  float rstd = rsqrtf(var + LNEPSc);
  bf16_t* orow = xn + (size_t)tok * DINc;
  for (int d = lane; d < DINc; d += 32)
    orow[d] = (bf16_t)((row[d] - mean) * rstd * g[d] + beta[d]);
}

// =====================================================================
// K1: GEMM out[m,n] = xn[m,:] . Wb[n,:] + bias[n]      (bf16 weights)
// Block tile 128x128, K-step 32, 8 waves (4x2), wave tile 32x64.
// A/B tiles DMA'd Global->LDS by the Tensor Data Mover (double-buffered,
// TENSORcnt-pipelined); falls back to manual staging without TDM.
// =====================================================================
__global__ __launch_bounds__(256) void kv_gemm_kernel(
    const bf16_t* __restrict__ xn, const bf16_t* __restrict__ Wb,
    const float* __restrict__ bias, bf16_t* __restrict__ out) {
  __shared__ __align__(16) bf16_t ldsA[2][128*32];   // [m][k] double-buffered
  __shared__ __align__(16) bf16_t ldsB[2][128*32];   // [n][k] double-buffered
  const int tid = threadIdx.x;
  const int wave = tid >> 5, lane = tid & 31;
  const int half = (lane >> 4) & 1, l16 = lane & 15;
  const int mBlk = blockIdx.x * 128;
  const int nBlk = blockIdx.y * 128;
  const int wm = wave >> 1, wn = wave & 1;
  const bf16_t* aBase = xn + (size_t)mBlk * DINc;
  const bf16_t* bBase = Wb + (size_t)nBlk * DINc;

  v8f acc[2][4];
  #pragma unroll
  for (int i = 0; i < 2; ++i)
    #pragma unroll
    for (int j = 0; j < 4; ++j)
      #pragma unroll
      for (int r = 0; r < 8; ++r) acc[i][j][r] = 0.f;

#ifdef HAVE_TDM
  if (wave == 0) tdm_load_tile(aBase, (unsigned)(uintptr_t)&ldsA[0][0]);
  if (wave == 1) tdm_load_tile(bBase, (unsigned)(uintptr_t)&ldsB[0][0]);
#endif

  const int KSTEPS = DINc / 32;   // 24
  for (int it = 0; it < KSTEPS; ++it) {
    const int buf = it & 1;
#ifdef HAVE_TDM
    if (it + 1 < KSTEPS) {
      if (wave == 0) {
        tdm_load_tile(aBase + (it + 1)*32, (unsigned)(uintptr_t)&ldsA[buf ^ 1][0]);
        __builtin_amdgcn_s_wait_tensorcnt(1);
      }
      if (wave == 1) {
        tdm_load_tile(bBase + (it + 1)*32, (unsigned)(uintptr_t)&ldsB[buf ^ 1][0]);
        __builtin_amdgcn_s_wait_tensorcnt(1);
      }
    } else {
      if (wave < 2) __builtin_amdgcn_s_wait_tensorcnt(0);
    }
#else
    // manual staging fallback (bf16 on both sides -> uint4 copies)
    const int k0 = it * 32;
    for (int i = tid; i < (128*32)/8; i += 256) {
      int m = i >> 2, kk = (i & 3) * 8;
      *(uint4*)&ldsA[buf][m*32 + kk] = *(const uint4*)&aBase[(size_t)m*DINc + k0 + kk];
      *(uint4*)&ldsB[buf][m*32 + kk] = *(const uint4*)&bBase[(size_t)m*DINc + k0 + kk];
    }
    if (k0 + 32 < DINc)
      __builtin_prefetch(&aBase[(size_t)(tid >> 1)*DINc + k0 + 32], 0, 0);
#endif
    __syncthreads();

    v16bf afrag[2], bfrag[4];
    #pragma unroll
    for (int i = 0; i < 2; ++i) {            // 16-bit A 16x32 layout
      int m = wm*32 + i*16 + l16;
      BF16x16 u;
      u.h[0] = *(const v8bf*)&ldsA[buf][m*32 + half*8];        // K 0-7 / 8-15
      u.h[1] = *(const v8bf*)&ldsA[buf][m*32 + 16 + half*8];   // K 16-23 / 24-31
      afrag[i] = u.v;
    }
    #pragma unroll
    for (int j = 0; j < 4; ++j) {            // 16-bit B 32x16 layout
      int n = wn*64 + j*16 + l16;
      BF16x16 u;
      u.h[0] = *(const v8bf*)&ldsB[buf][n*32 + half*8];        // K 0-7 / 8-15
      u.h[1] = *(const v8bf*)&ldsB[buf][n*32 + 16 + half*8];   // K 16-23 / 24-31
      bfrag[j] = u.v;
    }
    #pragma unroll
    for (int i = 0; i < 2; ++i)
      #pragma unroll
      for (int j = 0; j < 4; ++j)
        acc[i][j] = __builtin_amdgcn_wmma_f32_16x16x32_bf16(
            false, afrag[i], false, bfrag[j], (short)0, acc[i][j], false, false);
    __syncthreads();
  }
  // epilogue: + bias, bf16 store  (C layout: lane half -> M += 8)
  #pragma unroll
  for (int j = 0; j < 4; ++j) {
    int n = nBlk + wn*64 + j*16 + l16;
    float bv = bias[n];
    #pragma unroll
    for (int i = 0; i < 2; ++i)
      #pragma unroll
      for (int r = 0; r < 8; ++r) {
        int m = mBlk + wm*32 + i*16 + r + half*8;
        out[(size_t)m*Dc + n] = (bf16_t)(acc[i][j][r] + bv);
      }
  }
}

// =====================================================================
// K_vsum: vsum[b,d] = sum_j v[b,j,d]   (iteration-invariant)
// =====================================================================
__global__ __launch_bounds__(256) void vsum_kernel(
    const bf16_t* __restrict__ v, float* __restrict__ vsum) {
  int col = blockIdx.x * 256 + threadIdx.x;          // 0 .. B*D-1
  int b = col / Dc, d = col % Dc;
  float s = 0.f;
  const bf16_t* p = v + (size_t)b*Nc*Dc + d;
  for (int j = 0; j < Nc; ++j) s += (float)p[(size_t)j*Dc];
  vsum[col] = s;
}

// =====================================================================
// K_zero: clear attention accumulators (upd_acc + rowsum, contiguous)
// =====================================================================
__global__ __launch_bounds__(256) void zero_kernel(float* __restrict__ p, int n) {
  int i = blockIdx.x * 256 + threadIdx.x;
  if (i < n) p[i] = 0.f;
}

// =====================================================================
// K2: sn = LN(slots); q = sn @ Wq^T + bq -> bf16. One block per row.
// =====================================================================
__global__ __launch_bounds__(256) void slot_q_kernel(
    const float* __restrict__ slots, const float* __restrict__ g,
    const float* __restrict__ beta, const float* __restrict__ Wq,
    const float* __restrict__ bq, bf16_t* __restrict__ q) {
  __shared__ float xrow[Dc];
  __shared__ float red[2];
  int row = blockIdx.x, tid = threadIdx.x;
  if (tid < 2) red[tid] = 0.f;
  __syncthreads();
  float ps = 0.f, pq = 0.f;
  for (int d = tid; d < Dc; d += 256) {
    float v = slots[(size_t)row*Dc + d];
    xrow[d] = v; ps += v; pq += v*v;
  }
  atomicAdd(&red[0], ps); atomicAdd(&red[1], pq);
  __syncthreads();
  float mean = red[0] * (1.f/Dc);
  float var  = red[1] * (1.f/Dc) - mean*mean;
  float rstd = rsqrtf(var + LNEPSc);
  for (int d = tid; d < Dc; d += 256)
    xrow[d] = (xrow[d] - mean) * rstd * g[d] + beta[d];
  __syncthreads();
  for (int n = tid; n < Dc; n += 256) {
    float acc = bq[n];
    const float* w = &Wq[(size_t)n*Dc];
    for (int d = 0; d < Dc; ++d) acc += xrow[d] * w[d];
    q[(size_t)row*Dc + n] = (bf16_t)acc;
  }
}

// =====================================================================
// K3: attention.  grid = (N/512, H, B), block 256 (8 waves x 64 tokens).
//   dots = q.k (WMMA), per-lane softmax over 8 slot components of the
//   C tile, unnormalized upd & rowsum accumulated via global atomics.
// =====================================================================
__global__ __launch_bounds__(256) void attn_kernel(
    const bf16_t* __restrict__ q, const bf16_t* __restrict__ k,
    const bf16_t* __restrict__ v, float* __restrict__ upd_acc,
    float* __restrict__ rowsum) {
  __shared__ __align__(16) bf16_t ldsQ[16*64];        // [slot16][feat64]
  __shared__ __align__(16) bf16_t ldsAttn[8][16*64];  // per-wave [slot16][tok64]
  __shared__ __align__(16) bf16_t ldsVT[8][64*64];    // per-wave [d64][tok64]

  const int tid = threadIdx.x;
  const int wave = tid >> 5, lane = tid & 31;
  const int half = (lane >> 4) & 1, l16 = lane & 15;
  const int hIdx = blockIdx.y, b = blockIdx.z;
  const int jChunk = blockIdx.x * 512;
  const int jw = jChunk + wave * 64;

  // stage q tile (slots padded 8->16 with zeros)
  for (int i = tid; i < 16*64; i += 256) {
    int m = i >> 6, d = i & 63;
    ldsQ[i] = (m < Sc) ? q[((size_t)(b*Sc + m))*Dc + hIdx*HDc + d] : (bf16_t)0.f;
  }
  __syncthreads();

  v16bf afragQ[2];
  #pragma unroll
  for (int kb = 0; kb < 2; ++kb) {
    BF16x16 u;
    u.h[0] = *(const v8bf*)&ldsQ[l16*64 + kb*32 + half*8];
    u.h[1] = *(const v8bf*)&ldsQ[l16*64 + kb*32 + 16 + half*8];
    afragQ[kb] = u.v;
  }

  // stage V transposed into per-wave LDS: ldsVT[d][t]
  for (int i = lane; i < 64*8; i += 32) {
    int t = i >> 3, dg = (i & 7) * 8;
    v8bf vv = *(const v8bf*)&v[((size_t)(b*Nc + jw + t))*Dc + hIdx*HDc + dg];
    #pragma unroll
    for (int e = 0; e < 8; ++e) ldsVT[wave][(dg + e)*64 + t] = vv[e];
  }

  float rs[8];
  #pragma unroll
  for (int r = 0; r < 8; ++r) rs[r] = 0.f;

  // ---- dots + softmax over slot axis, 4 tiles of 16 tokens ----
  #pragma unroll
  for (int t4 = 0; t4 < 4; ++t4) {
    int jcol = jw + t4*16 + l16;
    v16bf bk0, bk1;
    {
      const bf16_t* kp = &k[((size_t)(b*Nc + jcol))*Dc + hIdx*HDc];
      BF16x16 u0, u1;
      u0.h[0] = *(const v8bf*)&kp[0  + half*8];
      u0.h[1] = *(const v8bf*)&kp[16 + half*8];
      u1.h[0] = *(const v8bf*)&kp[32 + half*8];
      u1.h[1] = *(const v8bf*)&kp[48 + half*8];
      bk0 = u0.v; bk1 = u1.v;
    }
    v8f dacc;
    #pragma unroll
    for (int r = 0; r < 8; ++r) dacc[r] = 0.f;
    dacc = __builtin_amdgcn_wmma_f32_16x16x32_bf16(false, afragQ[0], false, bk0,
                                                   (short)0, dacc, false, false);
    dacc = __builtin_amdgcn_wmma_f32_16x16x32_bf16(false, afragQ[1], false, bk1,
                                                   (short)0, dacc, false, false);
    // softmax over the 8 slot values (lane-local, lanes>=16 are padding)
    float a[8], mx = -1e30f;
    #pragma unroll
    for (int r = 0; r < 8; ++r) { a[r] = dacc[r] * SCALEc; mx = fmaxf(mx, a[r]); }
    float ssum = 0.f;
    #pragma unroll
    for (int r = 0; r < 8; ++r) { a[r] = __expf(a[r] - mx); ssum += a[r]; }
    float inv = 1.f / ssum;
    #pragma unroll
    for (int r = 0; r < 8; ++r) {
      a[r] = (half == 0) ? a[r] * inv : 0.f;
      rs[r] += a[r];
      ldsAttn[wave][(r + half*8)*64 + t4*16 + l16] = (bf16_t)a[r];
    }
  }

  // ---- upd = attn @ v : A from ldsAttn, B from ldsVT ----
  v8f accU[4];
  #pragma unroll
  for (int j = 0; j < 4; ++j)
    #pragma unroll
    for (int r = 0; r < 8; ++r) accU[j][r] = 0.f;

  #pragma unroll
  for (int kk = 0; kk < 2; ++kk) {
    BF16x16 ua;
    ua.h[0] = *(const v8bf*)&ldsAttn[wave][l16*64 + kk*32 + half*8];
    ua.h[1] = *(const v8bf*)&ldsAttn[wave][l16*64 + kk*32 + 16 + half*8];
    #pragma unroll
    for (int dt = 0; dt < 4; ++dt) {
      int d = dt*16 + l16;
      BF16x16 ub;
      ub.h[0] = *(const v8bf*)&ldsVT[wave][d*64 + kk*32 + half*8];
      ub.h[1] = *(const v8bf*)&ldsVT[wave][d*64 + kk*32 + 16 + half*8];
      accU[dt] = __builtin_amdgcn_wmma_f32_16x16x32_bf16(
          false, ua.v, false, ub.v, (short)0, accU[dt], false, false);
    }
  }

  // accumulate (slots 0..7 live in lanes 0..15)
  if (half == 0) {
    #pragma unroll
    for (int dt = 0; dt < 4; ++dt)
      #pragma unroll
      for (int r = 0; r < 8; ++r)
        atomicAdd(&upd_acc[(((size_t)(b*Hc + hIdx)*Sc + r)*HDc) + dt*16 + l16],
                  accU[dt][r]);
  }
  #pragma unroll
  for (int r = 0; r < 8; ++r)
    for (int m = 16; m >= 1; m >>= 1) rs[r] += __shfl_xor(rs[r], m, 32);
  if (lane == 0) {
    #pragma unroll
    for (int r = 0; r < 8; ++r)
      atomicAdd(&rowsum[(size_t)(b*Hc + hIdx)*Sc + r], rs[r]);
  }
}

// =====================================================================
// K4: finalize attention + Wo + GRU. One block per (b,s) row.
// =====================================================================
__global__ __launch_bounds__(256) void gru_kernel(
    const float* __restrict__ upd_acc, const float* __restrict__ rowsum,
    const float* __restrict__ vsum,
    const float* __restrict__ Wo, const float* __restrict__ bo,
    const float* __restrict__ Wih, const float* __restrict__ bih,
    const float* __restrict__ Whh, const float* __restrict__ bhh,
    float* __restrict__ slots) {
  __shared__ float updRow[Dc];
  __shared__ float oRow[Dc];
  __shared__ float hRow[Dc];
  int row = blockIdx.x, tid = threadIdx.x;
  int b = row / Sc, s = row % Sc;
  for (int d = tid; d < Dc; d += 256) {
    int h = d >> 6, hd = d & 63;
    float num = upd_acc[(((size_t)(b*Hc + h)*Sc + s)*HDc) + hd]
                + EPSc * vsum[(size_t)b*Dc + d];
    float den = rowsum[(size_t)(b*Hc + h)*Sc + s] + (float)Nc * EPSc;
    updRow[d] = num / den;
    hRow[d]   = slots[(size_t)row*Dc + d];
  }
  __syncthreads();
  for (int n = tid; n < Dc; n += 256) {
    float acc = bo[n];
    const float* w = &Wo[(size_t)n*Dc];
    for (int d = 0; d < Dc; ++d) acc += updRow[d] * w[d];
    oRow[n] = acc;
  }
  __syncthreads();
  for (int n = tid; n < Dc; n += 256) {
    float gir = bih[n], giz = bih[Dc+n], gin = bih[2*Dc+n];
    float ghr = bhh[n], ghz = bhh[Dc+n], ghn = bhh[2*Dc+n];
    const float* wir = &Wih[(size_t)n*Dc];
    const float* wiz = &Wih[(size_t)(Dc+n)*Dc];
    const float* win = &Wih[(size_t)(2*Dc+n)*Dc];
    const float* whr = &Whh[(size_t)n*Dc];
    const float* whz = &Whh[(size_t)(Dc+n)*Dc];
    const float* whn = &Whh[(size_t)(2*Dc+n)*Dc];
    for (int d = 0; d < Dc; ++d) {
      float xg = oRow[d], hh = hRow[d];
      gir += xg*wir[d]; giz += xg*wiz[d]; gin += xg*win[d];
      ghr += hh*whr[d]; ghz += hh*whz[d]; ghn += hh*whn[d];
    }
    float r = 1.f / (1.f + __expf(-(gir + ghr)));
    float z = 1.f / (1.f + __expf(-(giz + ghz)));
    float nn = tanhf(gin + r * ghn);
    slots[(size_t)row*Dc + n] = (1.f - z) * nn + z * hRow[n];
  }
}

// =====================================================================
// K5: coupling. One block per batch b.
// =====================================================================
__global__ __launch_bounds__(256) void coupling_kernel(
    const float* __restrict__ g, const float* __restrict__ beta,
    const float* __restrict__ Wsig, const float* __restrict__ bsig,
    const float* __restrict__ Wtr,  const float* __restrict__ btr,
    float* __restrict__ slots) {
  __shared__ float y[Sc*Dc];
  __shared__ float cfpre[Sc*Dc];
  __shared__ float fw[Sc*Sc];
  int b = blockIdx.x, tid = threadIdx.x;
  int wv = tid >> 5, lane = tid & 31;
  // LN per slot row (wave wv -> row wv)
  float ps = 0.f, pq = 0.f;
  const float* srow = &slots[(size_t)(b*Sc + wv)*Dc];
  for (int d = lane; d < Dc; d += 32) { float x = srow[d]; ps += x; pq += x*x; }
  for (int m = 16; m >= 1; m >>= 1) { ps += __shfl_xor(ps, m, 32); pq += __shfl_xor(pq, m, 32); }
  float mean = ps * (1.f/Dc);
  float var  = pq * (1.f/Dc) - mean*mean;
  float rstd = rsqrtf(var + LNEPSc);
  for (int d = lane; d < Dc; d += 32)
    y[wv*Dc + d] = (srow[d] - mean) * rstd * g[d] + beta[d];
  __syncthreads();
  if (tid < Sc*Sc) {
    int i = tid >> 3, j = tid & 7;
    float acc = bsig[j];
    const float* w = &Wsig[(size_t)j*Dc];
    for (int d = 0; d < Dc; ++d) acc += y[i*Dc + d] * w[d];
    fw[tid] = 1.f / (1.f + __expf(-acc));
  }
  __syncthreads();
  for (int idx = tid; idx < Sc*Dc; idx += 256) {
    int i = idx / Dc, d = idx % Dc;
    float s = 0.f;
    #pragma unroll
    for (int j = 0; j < Sc; ++j) s += fw[i*Sc + j] * y[j*Dc + d];
    cfpre[idx] = s;
  }
  __syncthreads();
  for (int idx = tid; idx < Sc*Dc; idx += 256) {
    int i = idx / Dc, n = idx % Dc;
    float acc = btr[n];
    const float* w = &Wtr[(size_t)n*Dc];
    for (int d = 0; d < Dc; ++d) acc += cfpre[i*Dc + d] * w[d];
    slots[(size_t)(b*Sc + i)*Dc + n] += acc;
  }
}

// =====================================================================
// host launcher
// =====================================================================
extern "C" void kernel_launch(void* const* d_in, const int* in_sizes, int n_in,
                              void* d_out, int out_size, void* d_ws, size_t ws_size,
                              hipStream_t stream) {
  const float* inputs     = (const float*)d_in[0];
  const float* slots_init = (const float*)d_in[1];
  const float* ln_in_g    = (const float*)d_in[2];
  const float* ln_in_b    = (const float*)d_in[3];
  const float* ln_s_g     = (const float*)d_in[4];
  const float* ln_s_b     = (const float*)d_in[5];
  const float* ln_ff_g    = (const float*)d_in[6];
  const float* ln_ff_b    = (const float*)d_in[7];
  const float* Wq = (const float*)d_in[8];   const float* bq = (const float*)d_in[9];
  const float* Wk = (const float*)d_in[10];  const float* bk = (const float*)d_in[11];
  const float* Wv = (const float*)d_in[12];  const float* bv = (const float*)d_in[13];
  const float* Wo = (const float*)d_in[14];  const float* bo = (const float*)d_in[15];
  const float* Wih = (const float*)d_in[16]; const float* bih = (const float*)d_in[17];
  const float* Whh = (const float*)d_in[18]; const float* bhh = (const float*)d_in[19];
  const float* Wtr = (const float*)d_in[20]; const float* btr = (const float*)d_in[21];
  const float* Wsig = (const float*)d_in[22];const float* bsig = (const float*)d_in[23];

  char* ws = (char*)d_ws;
  bf16_t* xn   = (bf16_t*)(ws + OFF_XN);
  bf16_t* kbuf = (bf16_t*)(ws + OFF_K);
  bf16_t* vbuf = (bf16_t*)(ws + OFF_V);
  bf16_t* qbuf = (bf16_t*)(ws + OFF_Q);
  float*  upd  = (float*) (ws + OFF_UPD);
  float*  rsum = (float*) (ws + OFF_RS);
  float*  vsum = (float*) (ws + OFF_VSUM);
  bf16_t* wkb  = (bf16_t*)(ws + OFF_WKB);
  bf16_t* wvb  = (bf16_t*)(ws + OFF_WVB);
  float*  slots = (float*)d_out;

  // init slots + one-shot weight conversion
  init_slots_kernel<<<(Bc*Sc*Dc + 255)/256, 256, 0, stream>>>(slots_init, slots);
  wcvt_kernel<<<(Dc*DINc + 255)/256, 256, 0, stream>>>(Wk, wkb, Dc*DINc);
  wcvt_kernel<<<(Dc*DINc + 255)/256, 256, 0, stream>>>(Wv, wvb, Dc*DINc);
  // LN(inputs) -> bf16
  ln_in_kernel<<<BN/8, 256, 0, stream>>>(inputs, ln_in_g, ln_in_b, xn);
  // K / V projections (WMMA GEMMs, TDM-staged tiles)
  {
    dim3 grid(BN/128, Dc/128);
    kv_gemm_kernel<<<grid, 256, 0, stream>>>(xn, wkb, bk, kbuf);
    kv_gemm_kernel<<<grid, 256, 0, stream>>>(xn, wvb, bv, vbuf);
  }
  // iteration-invariant V column sums
  vsum_kernel<<<(Bc*Dc)/256, 256, 0, stream>>>(vbuf, vsum);

  const int accN = Bc*Hc*Sc*HDc + Bc*Hc*Sc;   // upd_acc + rowsum, contiguous
  for (int it = 0; it < ITERSc; ++it) {
    zero_kernel<<<(accN + 255)/256, 256, 0, stream>>>(upd, accN);
    slot_q_kernel<<<Bc*Sc, 256, 0, stream>>>(slots, ln_s_g, ln_s_b, Wq, bq, qbuf);
    {
      dim3 grid(Nc/512, Hc, Bc);
      attn_kernel<<<grid, 256, 0, stream>>>(qbuf, kbuf, vbuf, upd, rsum);
    }
    gru_kernel<<<Bc*Sc, 256, 0, stream>>>(upd, rsum, vsum, Wo, bo,
                                          Wih, bih, Whh, bhh, slots);
    coupling_kernel<<<Bc, 256, 0, stream>>>(ln_ff_g, ln_ff_b, Wsig, bsig,
                                            Wtr, btr, slots);
  }
}
